// QLSTM_65481071409478
// MI455X (gfx1250) — matrix operand order repaired and checked by reading, not compiled, not run on hardware
//
#include <hip/hip_runtime.h>
#include <cstdint>

// ============================================================================
// QLSTM on MI455X (gfx1250), wave32 + WMMA, single persistent scan.
//
// The two parameterized circuit layers form a batch-independent 256x256
// complex unitary U per gate, precomputed once per launch. The data-encoding
// layer on |0..0> yields a product state psi0[k] = r[k]*phi[k] with phi[k] =
// (-i)^popcount(k) data-INdependent, so phi is folded into U at build time:
// amp = U' @ r with r purely real. Each scan step per gate is then:
//   angles = comb @ W^T  (f16 WMMA, K=96)
//   ampR/ampI = r @ U'^T (f16 WMMA, M=128 N=256 K=256, 2 real GEMMs)
//   Z = |amp|^2 @ SIGNS^T (f16 WMMA)
// 4 persistent workgroups (one gate each, one WGP each), device-scope atomic
// barrier once per step; U' planes live in L2 (192 MB, ~1 MB resident).
// All per-step transcendentals use hardware TRANS ops (v_cos/v_sin/v_exp/
// v_rcp) -- the angles are bounded, so no libm range reduction is needed.
// ============================================================================

typedef __attribute__((ext_vector_type(16))) _Float16 v16h;
typedef __attribute__((ext_vector_type(8)))  _Float16 h8;
typedef __attribute__((ext_vector_type(8)))  float    v8f;
typedef __attribute__((ext_vector_type(4)))  float    f4;

#define WMMA16(Af, Bf, Cf) \
  __builtin_amdgcn_wmma_f32_16x16x32_f16(false, (Af), false, (Bf), (short)0, (Cf), false, false)

// Fast transcendentals: single hardware TRANS instructions, NaN-free tails.
__device__ __forceinline__ float fast_sigmoid(float z) {
  return __builtin_amdgcn_rcpf(1.f + __expf(-z));   // z<<0: rcp(inf)=0; z>>0: 1
}
__device__ __forceinline__ float fast_tanh(float z) {
  // 1 - 2/(e^{2z}+1): e->inf => 1, e->0 => -1 (no inf*0 NaN)
  return 1.f - 2.f * __builtin_amdgcn_rcpf(__expf(2.f * z) + 1.f);
}

// ---------------- workspace layout ----------------
constexpr size_t U_PLANE = 65536;            // halfs per swizzled 256x256 plane
constexpr size_t U_GATE  = 2 * U_PLANE;      // planes: Re(U'), Im(U')
constexpr size_t WOFF    = 4 * U_GATE;       // weight frags [4][3 kb][32][16]
constexpr size_t SOFF    = WOFF + 4 * 1536;  // sign frags [8 kb][32][16]
constexpr size_t HALFTOT = SOFF + 4096;
constexpr size_t FBYTE   = HALFTOT * 2;      // float section byte offset
constexpr size_t BIASF   = 0;                // [4][8]
constexpr size_t GVF     = 32;               // [2 buf][4 gate][8 w][128 b]
constexpr size_t CNTBYTE = FBYTE + (GVF + 2 * 4 * 8 * 128) * 4;

// ---------------------------------------------------------------------------
__global__ void init_sync(unsigned* cnt) {
  if (threadIdx.x == 0 && blockIdx.x == 0) *cnt = 0u;
}

// ---------------------------------------------------------------------------
// Build U' per gate: butterfly-simulate basis column |k>, multiply by
// phi[k] = (-i)^popcount(k), scatter into f16 B-fragment-swizzled planes.
// grid = 4 gates * 256 columns, 256 threads (one amplitude j each).
// ---------------------------------------------------------------------------
__global__ void build_unitary(const float* fp, const float* ip, const float* up,
                              const float* op, _Float16* wsH) {
  const int g    = blockIdx.x >> 8;
  const int kcol = blockIdx.x & 255;
  const int t    = threadIdx.x;
  const float* params = (g == 0) ? fp : (g == 1) ? ip : (g == 2) ? up : op;

  __shared__ float sR[256], sI[256];
  sR[t] = (t == kcol) ? 1.f : 0.f;
  sI[t] = 0.f;
  __syncthreads();

  for (int d = 0; d < 2; ++d) {
    for (int w = 0; w < 8; ++w) {           // RX(theta) on wire w (bit 7-w)
      float th = params[d * 8 + w];
      float c = cosf(0.5f * th), s = sinf(0.5f * th);   // one-time: precise
      int pr = t ^ (1 << (7 - w));
      float aR = sR[t], aI = sI[t], bR = sR[pr], bI = sI[pr];
      __syncthreads();
      sR[t] = c * aR + s * bI;
      sI[t] = c * aI - s * bR;
      __syncthreads();
    }
    for (int w = 0; w < 7; ++w) {           // CNOT(w -> w+1)
      int cm = 1 << (7 - w), tm = 1 << (6 - w);
      int src = (t & cm) ? (t ^ tm) : t;
      float vR = sR[src], vI = sI[src];
      __syncthreads();
      sR[t] = vR; sI[t] = vI;
      __syncthreads();
    }
  }

  // fold phi[kcol] into the column
  float uR = sR[t], uI = sI[t], aR, aI;
  switch (__popc(kcol) & 3) {
    case 0:  aR =  uR; aI =  uI; break;
    case 1:  aR =  uI; aI = -uR; break;     // * (-i)
    case 2:  aR = -uR; aI = -uI; break;     // * (-1)
    default: aR = -uI; aI =  uR; break;     // * (+i)
  }
  // B-frag scatter: B[k][n=j]; k -> (kb, g2k, v, h); j -> (jt, lane%16)
  const int kb = kcol >> 5, g2k = (kcol >> 3) & 1, v = ((kcol >> 4) & 1) * 4 + ((kcol & 7) >> 1), h = kcol & 1;
  const int jt = t >> 4, lane = g2k * 16 + (t & 15);
  const size_t base = (size_t)g * U_GATE;
  const size_t idx  = (((size_t)kb * 16 + jt) * 32 + lane) * 16 + v * 2 + h;
  wsH[base + idx]           = (_Float16)aR;
  wsH[base + U_PLANE + idx] = (_Float16)aI;
}

// ---------------------------------------------------------------------------
// Pack weight (K=72 -> 96) and SIGNS^T B-fragments + biases.
// ---------------------------------------------------------------------------
__global__ void pack_aux(const float* fw, const float* fb, const float* iw,
                         const float* ib, const float* uw, const float* ub,
                         const float* ow, const float* ob,
                         _Float16* wsH, float* wsF) {
  const int t = threadIdx.x;
  if (blockIdx.x < 4) {
    const int g = blockIdx.x;
    const float* W  = (g == 0) ? fw : (g == 1) ? iw : (g == 2) ? uw : ow;
    const float* Bv = (g == 0) ? fb : (g == 1) ? ib : (g == 2) ? ub : ob;
    for (int i = t; i < 1536; i += 256) {
      int kb = i / 512, r = i % 512, lane = r / 16, vh = r % 16;
      int k = kb * 32 + 8 * (lane >> 4) + 16 * (vh >> 3) + (vh & 7);
      int n = lane & 15;
      float val = (n < 8 && k < 72) ? W[n * 72 + k] : 0.f;
      wsH[WOFF + (size_t)g * 1536 + i] = (_Float16)val;
    }
    if (t < 8) wsF[BIASF + g * 8 + t] = Bv[t];
  } else {
    for (int i = t; i < 4096; i += 256) {
      int kb = i / 512, r = i % 512, lane = r / 16, vh = r % 16;
      int j = kb * 32 + 8 * (lane >> 4) + 16 * (vh >> 3) + (vh & 7);
      int n = lane & 15;
      float val = (n < 8) ? (1.f - 2.f * (float)((j >> (7 - n)) & 1)) : 0.f;
      wsH[SOFF + i] = (_Float16)val;
    }
  }
}

// ---------------------------------------------------------------------------
// Persistent scan: grid = 4 blocks (gate f/i/u/o), 256 threads = 8 wave32.
// ---------------------------------------------------------------------------
__global__ void __launch_bounds__(256, 1)
qlstm_main(const float* __restrict__ x, const _Float16* __restrict__ wsH,
           const float* __restrict__ wsF, float* __restrict__ out,
           unsigned* cnt) {
  const int g    = blockIdx.x;
  const int tid  = threadIdx.x;
  const int lane = tid & 31;
  const int wv   = tid >> 5;       // wave id = batch tile
  const int m    = lane & 15;
  const int g2   = lane >> 4;
  const int bt   = wv;
  const int b    = bt * 16 + m;    // A-fragment batch row
  const int rbase = bt * 16 + 8 * g2;  // contiguous batch run for D rows

  __shared__ float    cS[8][128], sS[8][128];   // [wire][batch]
  __shared__ float    cxS[8][128];              // cell state [wire][batch]
  __shared__ _Float16 hxh[128][8];              // hidden state rows for GEMM1
  __shared__ _Float16 Pst[8][16 * 32];          // per-wave |amp|^2 staging

  for (int i = tid; i < 1024; i += 256) {
    cxS[i & 7][i >> 3] = 0.f;
    hxh[i >> 3][i & 7] = (_Float16)0.f;
  }
  __syncthreads();

  const _Float16* ufR = wsH + (size_t)g * U_GATE;
  const _Float16* ufI = ufR + U_PLANE;
  const _Float16* wf  = wsH + WOFF + (size_t)g * 1536;
  const _Float16* sf  = wsH + SOFF;
  const float bw = (m < 8) ? wsF[BIASF + g * 8 + m] : 0.f;
  float* gvbase = const_cast<float*>(wsF) + GVF;

  for (int s = 0; s < 32; ++s) {
    // ---- Phase A: angles = comb @ W^T + bias ----
    const float* xrow = x + ((size_t)s * 128 + b) * 64;
    if (s < 31) __builtin_prefetch(xrow + 128 * 64, 0, 0);
    v8f z = {0.f, 0.f, 0.f, 0.f, 0.f, 0.f, 0.f, 0.f};
#pragma unroll
    for (int kb = 0; kb < 2; ++kb) {            // K-blocks from x (contiguous runs)
      const f4* xp = (const f4*)(xrow + kb * 32 + 8 * g2);
      f4 a0 = xp[0], a1 = xp[1], a2 = xp[4], a3 = xp[5];
      v16h A = {};
#pragma unroll
      for (int q = 0; q < 4; ++q) {
        A[q]      = (_Float16)a0[q];
        A[4 + q]  = (_Float16)a1[q];
        A[8 + q]  = (_Float16)a2[q];
        A[12 + q] = (_Float16)a3[q];
      }
      v16h Bw = *(const v16h*)(wf + (size_t)kb * 512 + lane * 16);
      z = WMMA16(A, Bw, z);
    }
    {                                           // K-block 2: hx | zero pad
      h8 hrow = *(const h8*)(&hxh[b][0]);
      v16h A = {};
#pragma unroll
      for (int e = 0; e < 8; ++e) A[e] = g2 ? (_Float16)0.f : hrow[e];
      v16h Bw = *(const v16h*)(wf + 1024 + lane * 16);
      z = WMMA16(A, Bw, z);
    }
    if (m < 8) {                                // cos/sin of half angles (TRANS)
      f4 cva, cvb, sva, svb;
#pragma unroll
      for (int v = 0; v < 8; ++v) {
        float ha = 0.5f * (z[v] + bw);
        float cc = __cosf(ha), ss = __sinf(ha);
        if (v < 4) { cva[v] = cc; sva[v] = ss; }
        else       { cvb[v - 4] = cc; svb[v - 4] = ss; }
      }
      *(f4*)&cS[m][rbase] = cva; *(f4*)&cS[m][rbase + 4] = cvb;
      *(f4*)&sS[m][rbase] = sva; *(f4*)&sS[m][rbase + 4] = svb;
    }
    asm volatile("s_wait_dscnt 0" ::: "memory");

    // ---- Phase B: real product-state magnitudes r[k] -> A fragments ----
    float cb[8], sb[8];
#pragma unroll
    for (int w = 0; w < 8; ++w) { cb[w] = cS[w][b]; sb[w] = sS[w][b]; }
    // r(k) = rh[k>>4] * rl[k low bits];  k bit p <-> wire 7-p
    float rh[16];
#pragma unroll
    for (int i = 0; i < 16; ++i)
      rh[i] = ((i & 1) ? sb[3] : cb[3]) * ((i & 2) ? sb[2] : cb[2]) *
              ((i & 4) ? sb[1] : cb[1]) * ((i & 8) ? sb[0] : cb[0]);
    float sel4 = g2 ? sb[4] : cb[4];
    float rl[8];
#pragma unroll
    for (int u = 0; u < 8; ++u)
      rl[u] = ((u & 1) ? sb[7] : cb[7]) * ((u & 2) ? sb[6] : cb[6]) *
              ((u & 4) ? sb[5] : cb[5]) * sel4;
    v16h pA[8];
#pragma unroll
    for (int kb = 0; kb < 8; ++kb) {
      pA[kb] = (v16h){};
#pragma unroll
      for (int e = 0; e < 16; ++e)
        pA[kb][e] = (_Float16)(rh[kb * 2 + (e >> 3)] * rl[e & 7]);
    }

    // ---- amp = r @ U'^T (2 real GEMMs), Z = |amp|^2 @ SIGNS^T ----
    v8f zq = {0.f, 0.f, 0.f, 0.f, 0.f, 0.f, 0.f, 0.f};
    for (int jb = 0; jb < 8; ++jb) {
#pragma unroll
      for (int jt2 = 0; jt2 < 2; ++jt2) {
        const int jt = jb * 2 + jt2;
        v8f dR = {0.f, 0.f, 0.f, 0.f, 0.f, 0.f, 0.f, 0.f};
        v8f dI = {0.f, 0.f, 0.f, 0.f, 0.f, 0.f, 0.f, 0.f};
#pragma unroll
        for (int kb = 0; kb < 8; ++kb) {
          size_t off = (((size_t)kb * 16 + jt) * 32 + lane) * 16;
          v16h bR = *(const v16h*)(ufR + off);
          v16h bI = *(const v16h*)(ufI + off);
          dR = WMMA16(pA[kb], bR, dR);
          dI = WMMA16(pA[kb], bI, dI);
        }
#pragma unroll
        for (int v = 0; v < 8; ++v) {
          float prb = dR[v] * dR[v] + dI[v] * dI[v];
          Pst[wv][(v + 8 * g2) * 32 + jt2 * 16 + m] = (_Float16)prb;
        }
      }
      asm volatile("s_wait_dscnt 0" ::: "memory");
      h8 p0 = *(const h8*)&Pst[wv][m * 32 + 8 * g2];
      h8 p1 = *(const h8*)&Pst[wv][m * 32 + 16 + 8 * g2];
      v16h Pa = {};
#pragma unroll
      for (int e = 0; e < 8; ++e) { Pa[e] = p0[e]; Pa[8 + e] = p1[e]; }
      v16h Sb = *(const v16h*)(sf + (size_t)jb * 512 + lane * 16);
      zq = WMMA16(Pa, Sb, zq);
    }

    // ---- activation -> gate-value exchange (contiguous [w][b] runs) ----
    float* gvw = gvbase + ((size_t)((s & 1) * 4 + g)) * 1024 + m * 128;
    if (m < 8) {
      f4 ga, gb;
#pragma unroll
      for (int v = 0; v < 8; ++v) {
        float zz = zq[v];
        float val = (g == 2) ? fast_tanh(zz) : fast_sigmoid(zz);
        if (v < 4) ga[v] = val; else gb[v - 4] = val;
      }
      *(f4*)&gvw[rbase] = ga; *(f4*)&gvw[rbase + 4] = gb;
    }

    // ---- device-scope barrier across the 4 gate workgroups ----
    __syncthreads();
    __threadfence();
    if (tid == 0) {
      __hip_atomic_fetch_add(cnt, 1u, __ATOMIC_ACQ_REL, __HIP_MEMORY_SCOPE_AGENT);
      unsigned target = 4u * (unsigned)(s + 1);
      while (__hip_atomic_load(cnt, __ATOMIC_ACQUIRE, __HIP_MEMORY_SCOPE_AGENT) < target)
        __builtin_amdgcn_s_sleep(1);
    }
    __syncthreads();
    __threadfence();

    // ---- combine LSTM state (vectorized along contiguous batch runs) ----
    const float* gvr = gvbase + ((size_t)(s & 1)) * 4 * 1024 + m * 128 + rbase;
    if (m < 8) {
      f4 fva = *(const f4*)(gvr + 0 * 1024), fvb = *(const f4*)(gvr + 0 * 1024 + 4);
      f4 iva = *(const f4*)(gvr + 1 * 1024), ivb = *(const f4*)(gvr + 1 * 1024 + 4);
      f4 uva = *(const f4*)(gvr + 2 * 1024), uvb = *(const f4*)(gvr + 2 * 1024 + 4);
      f4 ova = *(const f4*)(gvr + 3 * 1024), ovb = *(const f4*)(gvr + 3 * 1024 + 4);
      f4 cxa = *(const f4*)&cxS[m][rbase], cxb = *(const f4*)&cxS[m][rbase + 4];
      f4 cna = fva * cxa + iva * uva;
      f4 cnb = fvb * cxb + ivb * uvb;
      *(f4*)&cxS[m][rbase] = cna; *(f4*)&cxS[m][rbase + 4] = cnb;
#pragma unroll
      for (int v = 0; v < 8; ++v) {
        float cn = (v < 4) ? cna[v] : cnb[v - 4];
        float ov = (v < 4) ? ova[v] : ovb[v - 4];
        float hv = ov * fast_tanh(cn);
        int bb = rbase + v;
        hxh[bb][m] = (_Float16)hv;
        if (g == 0) {
          out[((size_t)s * 128 + bb) * 8 + m] = hv;
          if (s == 31) {
            out[32768 + bb * 8 + m] = hv;   // final hx
            out[33792 + bb * 8 + m] = cn;   // final cx
          }
        }
      }
    }
    asm volatile("s_wait_dscnt 0" ::: "memory");
  }
}

// ---------------------------------------------------------------------------
extern "C" void kernel_launch(void* const* d_in, const int* in_sizes, int n_in,
                              void* d_out, int out_size, void* d_ws, size_t ws_size,
                              hipStream_t stream) {
  (void)in_sizes; (void)n_in; (void)out_size; (void)ws_size;
  const float* x  = (const float*)d_in[0];
  const float* fp = (const float*)d_in[1];
  const float* fw = (const float*)d_in[2];
  const float* fb = (const float*)d_in[3];
  const float* ip = (const float*)d_in[4];
  const float* iw = (const float*)d_in[5];
  const float* ib = (const float*)d_in[6];
  const float* up = (const float*)d_in[7];
  const float* uw = (const float*)d_in[8];
  const float* ub = (const float*)d_in[9];
  const float* op = (const float*)d_in[10];
  const float* ow = (const float*)d_in[11];
  const float* ob = (const float*)d_in[12];

  _Float16* wsH = (_Float16*)d_ws;
  float*    wsF = (float*)((char*)d_ws + FBYTE);
  unsigned* cnt = (unsigned*)((char*)d_ws + CNTBYTE);
  float*    out = (float*)d_out;

  hipLaunchKernelGGL(init_sync,     dim3(1),    dim3(32),  0, stream, cnt);
  hipLaunchKernelGGL(build_unitary, dim3(1024), dim3(256), 0, stream, fp, ip, up, op, wsH);
  hipLaunchKernelGGL(pack_aux,      dim3(5),    dim3(256), 0, stream,
                     fw, fb, iw, ib, uw, ub, ow, ob, wsH, wsF);
  hipLaunchKernelGGL(qlstm_main,    dim3(4),    dim3(256), 0, stream,
                     x, wsH, wsF, out, cnt);
}
